// CrossAttention_7103875908231
// MI455X (gfx1250) — matrix operand order repaired and checked by reading, not compiled, not run on hardware
//
#include <hip/hip_runtime.h>
#include <hip/hip_bf16.h>

typedef __bf16 bf16_t;
typedef __attribute__((ext_vector_type(16))) __bf16 v16bf;
typedef __attribute__((ext_vector_type(8)))  __bf16 v8bf;
typedef __attribute__((ext_vector_type(4)))  __bf16 v4bf;
typedef __attribute__((ext_vector_type(8)))  float  v8f;

constexpr int Bn = 8, Tn = 4096, Cn = 1024, Hn = 1024;

#define ASYNC_B128(l, g, off)                                                  \
  asm volatile("global_load_async_to_lds_b128 %0, %1, off offset:" #off        \
               :: "v"(l), "v"(g) : "memory")

// ---------------------------------------------------------------------------
// fp32 -> bf16 conversion, 4 elements/thread/iter (pure bandwidth)
// ---------------------------------------------------------------------------
__global__ __launch_bounds__(256)
void f32_to_bf16(const float4* __restrict__ in, v4bf* __restrict__ out, int n4) {
  for (int i = blockIdx.x * 256 + threadIdx.x; i < n4; i += gridDim.x * 256) {
    float4 f = in[i];
    v4bf o;
    o[0] = (bf16_t)f.x; o[1] = (bf16_t)f.y; o[2] = (bf16_t)f.z; o[3] = (bf16_t)f.w;
    out[i] = o;
  }
}

// ---------------------------------------------------------------------------
// 64x64 LDS-tiled bf16 transpose: out[c][r] = in[r][c], batched over z.
// ---------------------------------------------------------------------------
__global__ __launch_bounds__(256)
void transpose_bf16(const bf16_t* __restrict__ in, bf16_t* __restrict__ out,
                    int rows, int cols) {
  __shared__ bf16_t tile[64][72];
  const int bx = blockIdx.x * 64;            // col base (input)
  const int by = blockIdx.y * 64;            // row base (input)
  in  += (size_t)blockIdx.z * rows * cols;
  out += (size_t)blockIdx.z * rows * cols;
  const int t = threadIdx.x;
  #pragma unroll
  for (int j = 0; j < 2; ++j) {
    const int id = t + j * 256;              // 0..511
    const int r = id >> 3, c = (id & 7) * 8;
    *(uint4*)&tile[r][c] = *(const uint4*)&in[(size_t)(by + r) * cols + bx + c];
  }
  __syncthreads();
  #pragma unroll
  for (int j = 0; j < 2; ++j) {
    const int id = t + j * 256;
    const int r = id >> 3, c = (id & 7) * 8; // out row = bx+r, out col = by+c
    v8bf v;
    #pragma unroll
    for (int i = 0; i < 8; ++i) v[i] = tile[c + i][r];
    *(v8bf*)&out[(size_t)(bx + r) * rows + by + c] = v;
  }
}

// ---------------------------------------------------------------------------
// Tiled WMMA GEMM (NT):  Out[M,N] = scale * A[M,K] * B[N,K]^T
// 128x128 block tile, 128 threads (4 waves, 2x2), 64x64 per wave.
// K chunks of 64 (2 wmma k-steps); LDS double-buffered; both operands staged
// with global_load_async_to_lds_b128 (ASYNCcnt) overlapped with WMMA.
// ---------------------------------------------------------------------------
template<bool OUT_BF16>
__global__ __launch_bounds__(128)
void gemm_wmma(const bf16_t* __restrict__ A, int lda, long long sA,
               const bf16_t* __restrict__ Bm, int ldb, long long sB,
               void* __restrict__ Out, int ldo, long long sO,
               int K, float scale)
{
  constexpr int LSTR = 72;                  // 64 + 8 pad (row = 144B, 16B aligned)
  __shared__ bf16_t As[2][128 * LSTR];
  __shared__ bf16_t Bs[2][128 * LSTR];

  const int z = blockIdx.z;
  A  += (long long)z * sA;
  Bm += (long long)z * sB;

  const int rowBase = blockIdx.x * 128;
  const int colBase = blockIdx.y * 128;
  const int t    = threadIdx.x;
  const int lane = t & 31;
  const int wave = t >> 5;
  const int wm   = wave & 1;                // 0..1 -> 64-row slice
  const int wn   = wave >> 1;               // 0..1 -> 64-col slice

  // --- async-stage one 128x64 k-chunk of A and B into buffer `buf` ---------
  auto stage = [&](int kb, int buf) {
    const bf16_t* ga = A  + (size_t)(rowBase + t) * lda + kb;  // thread t = row t
    const bf16_t* gb = Bm + (size_t)(colBase + t) * ldb + kb;
    unsigned la = (unsigned)(size_t)&As[buf][t * LSTR];
    unsigned lb = (unsigned)(size_t)&Bs[buf][t * LSTR];
    ASYNC_B128(la, ga, 0);   ASYNC_B128(la, ga, 16);
    ASYNC_B128(la, ga, 32);  ASYNC_B128(la, ga, 48);
    ASYNC_B128(la, ga, 64);  ASYNC_B128(la, ga, 80);
    ASYNC_B128(la, ga, 96);  ASYNC_B128(la, ga, 112);
    ASYNC_B128(lb, gb, 0);   ASYNC_B128(lb, gb, 16);
    ASYNC_B128(lb, gb, 32);  ASYNC_B128(lb, gb, 48);
    ASYNC_B128(lb, gb, 64);  ASYNC_B128(lb, gb, 80);
    ASYNC_B128(lb, gb, 96);  ASYNC_B128(lb, gb, 112);
  };

  v8f acc[4][4] = {};
  const int nk = K / 64;

  stage(0, 0);
  for (int kc = 0; kc < nk; ++kc) {
    const int cur = kc & 1;
    if (kc + 1 < nk) {
      stage((kc + 1) * 64, cur ^ 1);
      asm volatile("s_wait_asynccnt 16" ::: "memory");  // current chunk landed
    } else {
      asm volatile("s_wait_asynccnt 0" ::: "memory");
    }
    __syncthreads();   // landed for all waves

    #pragma unroll
    for (int ks = 0; ks < 64; ks += 32) {
      // A frags: ISA 16-bit A layout (lanes 0-15: K{0-7,16-23}; 16-31: +8)
      v16bf afrag[4];
      #pragma unroll
      for (int mi = 0; mi < 4; ++mi) {
        const int r  = wm * 64 + mi * 16 + (lane & 15);
        const int k0 = ks + ((lane < 16) ? 0 : 8);
        v8bf lo = *(const v8bf*)&As[cur][r * LSTR + k0];
        v8bf hi = *(const v8bf*)&As[cur][r * LSTR + k0 + 16];
        #pragma unroll
        for (int i = 0; i < 8; ++i) { afrag[mi][i] = lo[i]; afrag[mi][i + 8] = hi[i]; }
      }
      // B frags: lane n = lane&15, same K order as A (lanes 0-15: K{0-7,16-23}; 16-31: +8)
      v16bf bfrag[4];
      #pragma unroll
      for (int ni = 0; ni < 4; ++ni) {
        const int n  = wn * 64 + ni * 16 + (lane & 15);
        const int k0 = ks + ((lane < 16) ? 0 : 8);
        v8bf b0 = *(const v8bf*)&Bs[cur][n * LSTR + k0];
        v8bf b1 = *(const v8bf*)&Bs[cur][n * LSTR + k0 + 16];
        #pragma unroll
        for (int i = 0; i < 8; ++i) { bfrag[ni][i] = b0[i]; bfrag[ni][i + 8] = b1[i]; }
      }
      // 16 WMMAs per k-step, 32 per barrier pair
      #pragma unroll
      for (int mi = 0; mi < 4; ++mi)
        #pragma unroll
        for (int ni = 0; ni < 4; ++ni)
          acc[mi][ni] = __builtin_amdgcn_wmma_f32_16x16x32_bf16(
              false, afrag[mi], false, bfrag[ni],
              (short)0, acc[mi][ni], false, false);
    }
    __syncthreads();   // safe to overwrite buffer `cur` next iteration
  }

  // ---- epilogue: C/D layout M = v + 8*(lane>=16), N = lane&15 -------------
  const int rOff = 8 * (lane >> 4);
  const int cOff = lane & 15;
  #pragma unroll
  for (int mi = 0; mi < 4; ++mi)
    #pragma unroll
    for (int ni = 0; ni < 4; ++ni)
      #pragma unroll
      for (int v = 0; v < 8; ++v) {
        const size_t rr = (size_t)rowBase + wm * 64 + mi * 16 + rOff + v;
        const size_t cc = (size_t)colBase + wn * 64 + ni * 16 + cOff;
        const float val = acc[mi][ni][v] * scale;
        if (OUT_BF16)
          ((bf16_t*)Out)[(long long)z * sO + rr * ldo + cc] = (bf16_t)val;
        else
          ((float*)Out)[(long long)z * sO + rr * ldo + cc] = val;
      }
}

// ---------------------------------------------------------------------------
// Row softmax, single pass over global data: Tn/256 = 16 values cached in
// registers; S read once, P written once.
// ---------------------------------------------------------------------------
__global__ __launch_bounds__(256)
void softmax_row(const float* __restrict__ S, bf16_t* __restrict__ P) {
  const long long row = blockIdx.x;
  const float* s = S + row * (long long)Tn;
  bf16_t*      p = P + row * (long long)Tn;
  const int t = threadIdx.x, lane = t & 31, wave = t >> 5;
  __shared__ float red[8];
  constexpr int NV = Tn / 256;   // 16

  float v[NV];
  #pragma unroll
  for (int j = 0; j < NV; ++j) v[j] = s[t + j * 256];

  float m = v[0];
  #pragma unroll
  for (int j = 1; j < NV; ++j) m = fmaxf(m, v[j]);
  #pragma unroll
  for (int o = 16; o > 0; o >>= 1) m = fmaxf(m, __shfl_xor(m, o, 32));
  if (lane == 0) red[wave] = m;
  __syncthreads();
  #pragma unroll
  for (int i = 0; i < 8; ++i) m = fmaxf(m, red[i]);
  __syncthreads();

  float sum = 0.f;
  #pragma unroll
  for (int j = 0; j < NV; ++j) { v[j] = __expf(v[j] - m); sum += v[j]; }
  #pragma unroll
  for (int o = 16; o > 0; o >>= 1) sum += __shfl_xor(sum, o, 32);
  if (lane == 0) red[wave] = sum;
  __syncthreads();
  float tot = 0.f;
  #pragma unroll
  for (int i = 0; i < 8; ++i) tot += red[i];
  const float inv = 1.0f / tot;

  #pragma unroll
  for (int j = 0; j < NV; ++j) p[t + j * 256] = (bf16_t)(v[j] * inv);
}

// ---------------------------------------------------------------------------
// Orchestration
// ---------------------------------------------------------------------------
extern "C" void kernel_launch(void* const* d_in, const int* in_sizes, int n_in,
                              void* d_out, int out_size, void* d_ws, size_t ws_size,
                              hipStream_t stream) {
  (void)in_sizes; (void)n_in; (void)out_size; (void)ws_size;
  const float* x  = (const float*)d_in[0];
  const float* Wq = (const float*)d_in[1];
  const float* Wk = (const float*)d_in[2];
  const float* Wv = (const float*)d_in[3];

  char* ws = (char*)d_ws;
  size_t off = 0;
  auto wsalloc = [&](size_t bytes) -> void* {
    void* pp = ws + off;
    off = (off + bytes + 255) & ~(size_t)255;
    return pp;
  };
  bf16_t* xb  = (bf16_t*)wsalloc((size_t)Bn * Tn * Cn * 2);   //  67 MB
  bf16_t* wqb = (bf16_t*)wsalloc((size_t)Hn * Cn * 2);        //   2 MB
  bf16_t* wkb = (bf16_t*)wsalloc((size_t)Hn * Cn * 2);
  bf16_t* wvb = (bf16_t*)wsalloc((size_t)Hn * Cn * 2);
  bf16_t* qb  = (bf16_t*)wsalloc((size_t)Bn * Tn * Hn * 2);   //  67 MB each
  bf16_t* kb  = (bf16_t*)wsalloc((size_t)Bn * Tn * Hn * 2);
  bf16_t* vb  = (bf16_t*)wsalloc((size_t)Bn * Tn * Hn * 2);
  bf16_t* vtb = (bf16_t*)wsalloc((size_t)Bn * Hn * Tn * 2);   //  67 MB (V^T)
  float*  Sf  = (float*) wsalloc((size_t)Bn * Tn * Tn * 4);   // 537 MB
  bf16_t* Pb  = (bf16_t*)wsalloc((size_t)Bn * Tn * Tn * 2);   // 268 MB

  // 1) fp32 -> bf16
  f32_to_bf16<<<2048, 256, 0, stream>>>((const float4*)x,  (v4bf*)xb,  Bn * Tn * Cn / 4);
  f32_to_bf16<<<256,  256, 0, stream>>>((const float4*)Wq, (v4bf*)wqb, Hn * Cn / 4);
  f32_to_bf16<<<256,  256, 0, stream>>>((const float4*)Wk, (v4bf*)wkb, Hn * Cn / 4);
  f32_to_bf16<<<256,  256, 0, stream>>>((const float4*)Wv, (v4bf*)wvb, Hn * Cn / 4);

  const long long sX = (long long)Tn * Cn;
  const long long sQ = (long long)Tn * Hn;
  const long long sS = (long long)Tn * Tn;

  // 2) QKV projections: [T,C] x [H,C]^T -> bf16 [T,H], batched over B
  dim3 gQKV(Tn / 128, Hn / 128, Bn);
  gemm_wmma<true><<<gQKV, 128, 0, stream>>>(xb, Cn, sX, wqb, Cn, 0, qb, Hn, sQ, Cn, 1.0f);
  gemm_wmma<true><<<gQKV, 128, 0, stream>>>(xb, Cn, sX, wkb, Cn, 0, kb, Hn, sQ, Cn, 1.0f);
  gemm_wmma<true><<<gQKV, 128, 0, stream>>>(xb, Cn, sX, wvb, Cn, 0, vb, Hn, sQ, Cn, 1.0f);

  // 3) V^T so the P*V GEMM is NT with fully-async staging
  dim3 gT(Hn / 64, Tn / 64, Bn);
  transpose_bf16<<<gT, 256, 0, stream>>>(vb, vtb, Tn, Hn);

  // 4) scores: S = (Q K^T) / sqrt(C) -> fp32 [T,T]
  dim3 gS(Tn / 128, Tn / 128, Bn);
  gemm_wmma<false><<<gS, 128, 0, stream>>>(qb, Hn, sQ, kb, Hn, sQ, Sf, Tn, sS, Hn, 0.03125f);

  // 5) softmax rows -> P bf16 (single global pass)
  softmax_row<<<Bn * Tn, 256, 0, stream>>>(Sf, Pb);

  // 6) O = P V = P @ (V^T)^T -> fp32 out [T,H]
  dim3 gO(Tn / 128, Hn / 128, Bn);
  gemm_wmma<false><<<gO, 128, 0, stream>>>(Pb, Tn, sS, vtb, Tn, (long long)Hn * Tn,
                                           d_out, Hn, sQ, Tn, 1.0f);
}
